// RoPEMultiHeadAttention_61847529062905
// MI455X (gfx1250) — matrix operand order repaired and checked by reading, not compiled, not run on hardware
//
#include <hip/hip_runtime.h>
#include <hip/hip_bf16.h>
#include <stdint.h>

// ---------------------------------------------------------------------------
// RoPE GQA attention for gfx1250 (MI455X), wave32 + v_wmma_f32_16x16x32_f16.
// B=2, S=2048, D=2048, H=16, Hkv=4, Dh=128.
// Pipeline: cvt -> fused QKV GEMM (WMMA) -> RoPE in place -> flash attention
// (WMMA, online softmax) -> output GEMM (WMMA) -> f32 out.
// Software-pipelined (register-staged double-buffered LDS) in both GEMM and
// attention so next-tile global loads overlap the current tile's WMMAs.
// ---------------------------------------------------------------------------

typedef __attribute__((ext_vector_type(16))) _Float16 v16h;
typedef __attribute__((ext_vector_type(8)))  float    v8f;
typedef __attribute__((ext_vector_type(2)))  _Float16 h2;
typedef __attribute__((ext_vector_type(2)))  __fp16   fp16x2;   // cvt_pkrtz result

union FragH {            // one WMMA 16-bit operand: 8 VGPRs = 16 halfs / lane
  v16h h;
  uint32_t u[8];
  uint4 q[2];
};

union PackH { uint32_t w; h2 v; fp16x2 f; };

#define WMMA16(A, B, C) \
  __builtin_amdgcn_wmma_f32_16x16x32_f16(false, (A), false, (B), (short)0, (C), false, false)

// ---------------------------------------------------------------------------
// Elementwise conversion kernels
// ---------------------------------------------------------------------------
__global__ void cvt_f32_to_f16(const float* __restrict__ src,
                               _Float16* __restrict__ dst, int n) {
  int i = blockIdx.x * blockDim.x + threadIdx.x;
  int stride = gridDim.x * blockDim.x;
  for (; i < n; i += stride) dst[i] = (_Float16)src[i];
}

// Fuse Wq [2048x2048] | Wk [2048x512] | Wv [2048x512] -> f16 [2048x3072]
__global__ void build_wfused(const float* __restrict__ Wq,
                             const float* __restrict__ Wk,
                             const float* __restrict__ Wv,
                             _Float16* __restrict__ Wf) {
  int i = blockIdx.x * blockDim.x + threadIdx.x;
  int stride = gridDim.x * blockDim.x;
  const int total = 2048 * 3072;
  for (; i < total; i += stride) {
    int r = i / 3072, c = i % 3072;
    float v;
    if (c < 2048)      v = Wq[(size_t)r * 2048 + c];
    else if (c < 2560) v = Wk[(size_t)r * 512 + (c - 2048)];
    else               v = Wv[(size_t)r * 512 + (c - 2560)];
    Wf[i] = (_Float16)v;
  }
}

// ---------------------------------------------------------------------------
// RoPE applied in place on QKV f16 buffer [4096 rows x 3072].
// Q cols [0,2048): 16 heads x 64 pairs. K cols [2048,2560): 4 heads x 64 pairs.
// ---------------------------------------------------------------------------
__global__ void rope_inplace(_Float16* __restrict__ QKV) {
  const int total = 4096 * 1280;              // 1280 pairs per row
  int idx = blockIdx.x * blockDim.x + threadIdx.x;
  if (idx >= total) return;
  int row = idx / 1280, p = idx % 1280;
  int pos = row & 2047;                       // s within sequence
  int i, col0;
  if (p < 1024) { int hh = p >> 6; i = p & 63; col0 = hh * 128 + 2 * i; }
  else { int p2 = p - 1024; int hh = p2 >> 6; i = p2 & 63; col0 = 2048 + hh * 128 + 2 * i; }
  float freq = __expf(-(float)i * 0.14391156831f);   // ln(10000)/64
  float ang = (float)pos * freq;
  float sn, cs;
  __sincosf(ang, &sn, &cs);
  _Float16* ptr = QKV + (size_t)row * 3072 + col0;
  float x0 = (float)ptr[0], x1 = (float)ptr[1];
  ptr[0] = (_Float16)(x0 * cs - x1 * sn);
  ptr[1] = (_Float16)(x0 * sn + x1 * cs);
}

// ---------------------------------------------------------------------------
// Tiled f16 GEMM, f32 accumulate: C[MxN] = A[MxK] * B[KxN], all row-major.
// Block tile 128x128, K-tile 32, 8 waves (wave32), each wave: 2x4 WMMA tiles.
// Double-buffered LDS, register-staged prefetch of the next K-tile.
// LDS layouts chosen so A and B fragments load as 2 x ds_read_b128 each,
// matching the CDNA5 16-bit A(16x32)/B(32x16) wave32 VGPR layouts.
// ---------------------------------------------------------------------------
template <bool OUT16>
__global__ __launch_bounds__(256) void gemm_f16(const _Float16* __restrict__ A,
                                                const _Float16* __restrict__ B,
                                                void* __restrict__ Cout,
                                                int M, int N, int K) {
  constexpr int BM = 128, BN = 128, BK = 32, PAD = 8;
  __shared__ _Float16 As[2][BM][BK + PAD];   // row-major, stride 40 halfs
  __shared__ _Float16 Bt[2][BN][BK + PAD];   // [n][k]: k-contiguous per column

  const int tid = threadIdx.x;
  const int lane = tid & 31, wv = tid >> 5;
  const int wr = wv >> 1, wc = wv & 1;       // 4 row-waves x 2 col-waves
  const int l16 = lane >> 4, nn = lane & 15;
  const int mBase = blockIdx.y * BM, nBase = blockIdx.x * BN;
  const int nTiles = K / BK;

  // Per-thread cooperative-load geometry (2 x 16B segments for A and B).
  const int aRow0 = (tid * 2) >> 2, aCs0 = (tid * 2) & 3;
  const int aRow1 = (tid * 2 + 1) >> 2, aCs1 = (tid * 2 + 1) & 3;
  const int bKr0 = (tid * 2) >> 4, bCs0 = (tid * 2) & 15;
  const int bKr1 = (tid * 2 + 1) >> 4, bCs1 = (tid * 2 + 1) & 15;

  auto loadTile = [&](int k0, uint4* ra, uint4* rb) {
    ra[0] = *(const uint4*)(A + (size_t)(mBase + aRow0) * K + k0 + aCs0 * 8);
    ra[1] = *(const uint4*)(A + (size_t)(mBase + aRow1) * K + k0 + aCs1 * 8);
    rb[0] = *(const uint4*)(B + (size_t)(k0 + bKr0) * N + nBase + bCs0 * 8);
    rb[1] = *(const uint4*)(B + (size_t)(k0 + bKr1) * N + nBase + bCs1 * 8);
  };
  auto storeTile = [&](int buf, const uint4* ra, const uint4* rb) {
    *(uint4*)(&As[buf][aRow0][aCs0 * 8]) = ra[0];
    *(uint4*)(&As[buf][aRow1][aCs1 * 8]) = ra[1];
    const _Float16* h0 = (const _Float16*)&rb[0];
    const _Float16* h1 = (const _Float16*)&rb[1];
#pragma unroll
    for (int j = 0; j < 8; ++j) Bt[buf][bCs0 * 8 + j][bKr0] = h0[j];
#pragma unroll
    for (int j = 0; j < 8; ++j) Bt[buf][bCs1 * 8 + j][bKr1] = h1[j];
  };

  v8f acc[2][4] = {};

  {  // prologue: stage tile 0 into buffer 0
    uint4 ra[2], rb[2];
    loadTile(0, ra, rb);
    storeTile(0, ra, rb);
  }

  for (int it = 0; it < nTiles; ++it) {
    const int cur = it & 1;
    __syncthreads();

    uint4 ra[2], rb[2];
    const bool hasNext = (it + 1) < nTiles;
    if (hasNext) loadTile((it + 1) * BK, ra, rb);   // overlap with WMMAs below

    FragH af[2], bf[4];
#pragma unroll
    for (int rt = 0; rt < 2; ++rt) {          // A frag: lane = row, kb = 0/8
      int r = wr * 32 + rt * 16 + nn;
      int kb = l16 * 8;
      af[rt].q[0] = *(const uint4*)(&As[cur][r][kb]);
      af[rt].q[1] = *(const uint4*)(&As[cur][r][16 + kb]);
    }
#pragma unroll
    for (int ct = 0; ct < 4; ++ct) {          // B frag: lane = col, kb = 0/8
      int n = wc * 64 + ct * 16 + nn;
      int kb = l16 * 8;
      bf[ct].q[0] = *(const uint4*)(&Bt[cur][n][kb]);
      bf[ct].q[1] = *(const uint4*)(&Bt[cur][n][16 + kb]);
    }
#pragma unroll
    for (int rt = 0; rt < 2; ++rt)
#pragma unroll
      for (int ct = 0; ct < 4; ++ct)
        acc[rt][ct] = WMMA16(af[rt].h, bf[ct].h, acc[rt][ct]);

    if (hasNext) storeTile(cur ^ 1, ra, rb);
  }

  // Epilogue: C layout -> lane = col (nn), rows across VGPRs (+8 for hi half)
#pragma unroll
  for (int rt = 0; rt < 2; ++rt)
#pragma unroll
    for (int ct = 0; ct < 4; ++ct) {
      int rbase = mBase + wr * 32 + rt * 16 + l16 * 8;
      int col = nBase + wc * 64 + ct * 16 + nn;
#pragma unroll
      for (int r = 0; r < 8; ++r) {
        float v = acc[rt][ct][r];
        if constexpr (OUT16)
          ((_Float16*)Cout)[(size_t)(rbase + r) * N + col] = (_Float16)v;
        else
          ((float*)Cout)[(size_t)(rbase + r) * N + col] = v;
      }
    }
}

// ---------------------------------------------------------------------------
// Flash attention, scores computed transposed so lane == query.
// Grid: (S/128, B*H). Block: 256 threads = 8 waves, each wave owns 16 queries.
// Per 32-key tile per wave: 8 WMMA (S^T = K*Q^T) + 8 WMMA (O^T = V^T*P^T).
// 1/sqrt(Dh) folded into the Q fragments once; double-buffered K/V tiles.
// ---------------------------------------------------------------------------
__global__ __launch_bounds__(256) void attn_kernel(const _Float16* __restrict__ QKV,
                                                   _Float16* __restrict__ AO) {
  constexpr int S = 2048, LDQ = 3072, DH = 128;
  constexpr int K_OFF = 2048, V_OFF = 2560;
  __shared__ _Float16 Kt[2][32][DH + 8];   // [key][dim], stride 136
  __shared__ _Float16 Vt[2][DH][40];       // [dim][key], transposed V

  const int tid = threadIdx.x, lane = tid & 31, wv = tid >> 5;
  const int b = blockIdx.y >> 4, h = blockIdx.y & 15, hkv = h & 3;
  const int l16 = lane >> 4, nn = lane & 15;
  const size_t rowB = (size_t)b * S;
  const int qbase = blockIdx.x * 128 + wv * 16;

  // Preload Q as B-fragments (lane = query, k = head dim), 4 chunks of 32,
  // with 1/sqrt(128) folded in via packed f16 multiplies.
  FragH qf[4];
  {
    const _Float16 qs = (_Float16)0.08838834764831845f;
    const h2 qsv = {qs, qs};
    const _Float16* qp = QKV + (rowB + qbase + nn) * LDQ + h * DH;
#pragma unroll
    for (int c = 0; c < 4; ++c) {
      int kb = c * 32 + l16 * 8;
      qf[c].q[0] = *(const uint4*)(qp + kb);
      qf[c].q[1] = *(const uint4*)(qp + kb + 16);
#pragma unroll
      for (int i = 0; i < 8; ++i) {
        PackH t; t.w = qf[c].u[i];
        t.v *= qsv;
        qf[c].u[i] = t.w;
      }
    }
  }

  // Cooperative tile-load geometry (2 x 16B segments per thread).
  const int key0 = (tid * 2) >> 4, cs0 = (tid * 2) & 15;
  const int key1 = (tid * 2 + 1) >> 4, cs1 = (tid * 2 + 1) & 15;

  auto loadTile = [&](int kb0, uint4* rk, uint4* rv) {
    const size_t g0 = (rowB + kb0 + key0) * LDQ;
    const size_t g1 = (rowB + kb0 + key1) * LDQ;
    rk[0] = *(const uint4*)(QKV + g0 + K_OFF + hkv * DH + cs0 * 8);
    rk[1] = *(const uint4*)(QKV + g1 + K_OFF + hkv * DH + cs1 * 8);
    rv[0] = *(const uint4*)(QKV + g0 + V_OFF + hkv * DH + cs0 * 8);
    rv[1] = *(const uint4*)(QKV + g1 + V_OFF + hkv * DH + cs1 * 8);
  };
  auto storeTile = [&](int buf, const uint4* rk, const uint4* rv) {
    *(uint4*)(&Kt[buf][key0][cs0 * 8]) = rk[0];
    *(uint4*)(&Kt[buf][key1][cs1 * 8]) = rk[1];
    const _Float16* h0 = (const _Float16*)&rv[0];
    const _Float16* h1 = (const _Float16*)&rv[1];
#pragma unroll
    for (int j = 0; j < 8; ++j) Vt[buf][cs0 * 8 + j][key0] = h0[j];
#pragma unroll
    for (int j = 0; j < 8; ++j) Vt[buf][cs1 * 8 + j][key1] = h1[j];
  };

  v8f oacc[8] = {};                 // O^T accum: 8 dim-tiles x (lane = query)
  float runm = -1e30f, lsum = 0.f;

  {  // prologue: stage key-tile 0 into buffer 0
    uint4 rk[2], rv[2];
    loadTile(0, rk, rv);
    storeTile(0, rk, rv);
  }

  for (int it = 0; it < S / 32; ++it) {
    const int cur = it & 1;
    __syncthreads();

    uint4 rk[2], rv[2];
    const bool hasNext = (it + 1) < S / 32;
    if (hasNext) loadTile((it + 1) * 32, rk, rv);   // overlap with compute

    // ---- S^T[key][query] for 32 keys: two 16x16 C tiles (Q pre-scaled) ----
    v8f sc0 = {}, sc1 = {};
#pragma unroll
    for (int c = 0; c < 4; ++c) {
      int kb = c * 32 + l16 * 8;
      FragH ak0, ak1;                       // A frag: lane = key row
      ak0.q[0] = *(const uint4*)(&Kt[cur][nn][kb]);
      ak0.q[1] = *(const uint4*)(&Kt[cur][nn][kb + 16]);
      ak1.q[0] = *(const uint4*)(&Kt[cur][16 + nn][kb]);
      ak1.q[1] = *(const uint4*)(&Kt[cur][16 + nn][kb + 16]);
      sc0 = WMMA16(ak0.h, qf[c].h, sc0);
      sc1 = WMMA16(ak1.h, qf[c].h, sc1);
    }

    // ---- Online softmax. Each lane holds 16 scores of its query; the
    //      other 16 keys of this tile live in lane ^ 16 (same query). ----
    float p[16];
    float tmax = -1e30f;
#pragma unroll
    for (int j = 0; j < 8; ++j) { p[j] = sc0[j]; tmax = fmaxf(tmax, p[j]); }
#pragma unroll
    for (int j = 0; j < 8; ++j) { p[8 + j] = sc1[j]; tmax = fmaxf(tmax, p[8 + j]); }
    tmax = fmaxf(tmax, __shfl_xor(tmax, 16, 32));
    float newm = fmaxf(runm, tmax);
    float corr = __expf(runm - newm);
    runm = newm;
    float ps = 0.f;
#pragma unroll
    for (int j = 0; j < 16; ++j) { p[j] = __expf(p[j] - newm); ps += p[j]; }
    lsum = lsum * corr + ps;
#pragma unroll
    for (int t = 0; t < 8; ++t)
#pragma unroll
      for (int j = 0; j < 8; ++j) oacc[t][j] *= corr;

    // ---- Rebuild P^T as a B fragment (lane = query, k = key index).
    //      Lane holds keys {b0..b0+7, 16+b0..16+b0+7}, b0 = l16*8; B frag
    //      needs keys {kb..kb+15}, kb = l16*16 -> one xor-16 exchange. ----
    uint32_t pl[4], ph[4];
#pragma unroll
    for (int i = 0; i < 4; ++i) {
      PackH a; a.f = __builtin_amdgcn_cvt_pkrtz(p[2 * i], p[2 * i + 1]);
      pl[i] = a.w;
      PackH bpk; bpk.f = __builtin_amdgcn_cvt_pkrtz(p[8 + 2 * i], p[9 + 2 * i]);
      ph[i] = bpk.w;
    }
    FragH pf;
#pragma unroll
    for (int i = 0; i < 4; ++i) {
      uint32_t opl = (uint32_t)__shfl_xor((int)pl[i], 16, 32);
      uint32_t oph = (uint32_t)__shfl_xor((int)ph[i], 16, 32);
      pf.u[i]     = l16 ? oph : pl[i];   // keys kb+0..7
      pf.u[4 + i] = l16 ? ph[i] : opl;   // keys kb+8..15
    }

    // ---- O^T += V^T * P^T: 8 dim-tiles of 16 ----
#pragma unroll
    for (int dt = 0; dt < 8; ++dt) {
      int d = dt * 16 + nn;
      int kb2 = l16 * 16;
      FragH av;                            // A frag: lane = dim row, k = key
      av.q[0] = *(const uint4*)(&Vt[cur][d][kb2]);
      av.q[1] = *(const uint4*)(&Vt[cur][d][kb2 + 8]);
      oacc[dt] = WMMA16(av.h, pf.h, oacc[dt]);
    }

    if (hasNext) storeTile(cur ^ 1, rk, rv);
  }

  // Normalize and store: lane's column is query qbase+nn; 8 contiguous dims
  // per tile per lane -> one 16B store each.
  float ltot = lsum + __shfl_xor(lsum, 16, 32);
  float inv = 1.0f / ltot;
  _Float16* op = AO + (rowB + qbase + nn) * 2048 + h * DH + l16 * 8;
#pragma unroll
  for (int dt = 0; dt < 8; ++dt) {
    union { uint4 q; _Float16 hh[8]; } st;
#pragma unroll
    for (int j = 0; j < 8; ++j) st.hh[j] = (_Float16)(oacc[dt][j] * inv);
    *(uint4*)(op + dt * 16) = st.q;
  }
}

// ---------------------------------------------------------------------------
// Launch
// ---------------------------------------------------------------------------
extern "C" void kernel_launch(void* const* d_in, const int* in_sizes, int n_in,
                              void* d_out, int out_size, void* d_ws, size_t ws_size,
                              hipStream_t stream) {
  const float* x  = (const float*)d_in[0];
  const float* Wq = (const float*)d_in[1];
  const float* Wk = (const float*)d_in[2];
  const float* Wv = (const float*)d_in[3];
  const float* Wo = (const float*)d_in[4];
  float* out = (float*)d_out;

  char* ws = (char*)d_ws;
  _Float16* Xh   = (_Float16*)(ws);                      // 4096x2048 f16  16 MB
  _Float16* Wf   = (_Float16*)(ws + 16777216);           // 2048x3072 f16  12 MB
  _Float16* Woh  = (_Float16*)(ws + 29360128);           // 2048x2048 f16   8 MB
  _Float16* QKV  = (_Float16*)(ws + 37748736);           // 4096x3072 f16  24 MB
  _Float16* AO   = (_Float16*)(ws + 62914560);           // 4096x2048 f16  16 MB

  cvt_f32_to_f16<<<8192, 256, 0, stream>>>(x, Xh, 4096 * 2048);
  build_wfused<<<8192, 256, 0, stream>>>(Wq, Wk, Wv, Wf);
  cvt_f32_to_f16<<<8192, 256, 0, stream>>>(Wo, Woh, 2048 * 2048);

  // Fused QKV projection: [4096x2048] x [2048x3072] -> f16 QKV
  gemm_f16<true><<<dim3(24, 32), 256, 0, stream>>>(Xh, Wf, (void*)QKV, 4096, 3072, 2048);

  // RoPE on Q and K columns, in place
  rope_inplace<<<20480, 256, 0, stream>>>(QKV);

  // Attention: grid (S/128 q-blocks, B*H heads)
  attn_kernel<<<dim3(16, 32), 256, 0, stream>>>(QKV, AO);

  // Output projection: [4096x2048] x [2048x2048] -> f32 out
  gemm_f16<false><<<dim3(16, 32), 256, 0, stream>>>(AO, Woh, (void*)out, 4096, 2048, 2048);
}
